// SwiGLUMoE_14181982011877
// MI455X (gfx1250) — matrix operand order repaired and checked by reading, compiled, not run on hardware
//
#include <hip/hip_runtime.h>
#include <hip/hip_bf16.h>
#include <stdint.h>

typedef unsigned short u16;
typedef __attribute__((ext_vector_type(16))) __bf16 v16bf;
typedef __attribute__((ext_vector_type(8)))  float  v8f;

namespace {
constexpr int kB = 2, kS = 2048, kH = 896, kI = 4864, kE = 4;
constexpr int kTok = kB * kS;              // 4096 tokens
constexpr int MT   = 32;                   // tokens per block (2 WMMA M-tiles)
constexpr int ICH  = 128;                  // I-chunk width (8 WMMA N-tiles)
constexpr int XSTR = kH + 8;               // 904: padded x-tile stride (452 dw % 64 = 4 -> no bank conflicts)
constexpr int HSTR = ICH + 8;              // 136: padded h-tile stride (68 dw % 64 = 4)
constexpr size_t kWElems = (size_t)kE * (size_t)kI * (size_t)kH;  // 17,432,576
// workspace layout (all offsets 256B aligned by construction)
constexpr size_t OFF_GW = 0;
constexpr size_t OFF_UW = OFF_GW + kWElems * 2;
constexpr size_t OFF_DW = OFF_UW + kWElems * 2;
constexpr size_t OFF_WF = OFF_DW + kWElems * 2;
}

__device__ __forceinline__ u16 f2bf(float f) {
  // round-to-nearest-even f32 -> bf16
  unsigned int u = __float_as_uint(f);
  unsigned int r = u + 0x7FFFu + ((u >> 16) & 1u);
  return (u16)(r >> 16);
}

union FragU {
  v16bf v;
  uint4 q[2];
};

__device__ __forceinline__ v8f wmma_bf16(const FragU& a, const FragU& b, v8f c) {
  return __builtin_amdgcn_wmma_f32_16x16x32_bf16(
      /*neg_a=*/false, a.v, /*neg_b=*/false, b.v,
      /*c_mod=*/(short)0, c, /*reuse_a=*/false, /*reuse_b=*/false);
}

// ---------------------------------------------------------------- conversions
__global__ void cvt_f32_bf16(const float* __restrict__ s, u16* __restrict__ d,
                             size_t n) {
  size_t i = ((size_t)blockIdx.x * blockDim.x + threadIdx.x) * 4;
  size_t stride = (size_t)gridDim.x * blockDim.x * 4;
  for (; i < n; i += stride) {
    float4 f = *reinterpret_cast<const float4*>(s + i);
    u16 o[4] = {f2bf(f.x), f2bf(f.y), f2bf(f.z), f2bf(f.w)};
    *reinterpret_cast<uint2*>(d + i) = *reinterpret_cast<const uint2*>(o);
  }
}

// ---------------------------------------------------------------- router
// One wave (32 lanes) per token. fp32 throughout, matches reference semantics:
// softmax -> top-2 (first-index ties) -> renormalize -> dense [tok,E] weights.
__global__ void router_k(const float* __restrict__ x,
                         const float* __restrict__ rw,
                         float* __restrict__ wf) {
  const int wave = threadIdx.x >> 5;
  const int lane = threadIdx.x & 31;
  const long tok = (long)blockIdx.x * 4 + wave;
  if (tok >= kTok) return;

  const float* xr = x + (size_t)tok * kH;
  float a0 = 0.f, a1 = 0.f, a2 = 0.f, a3 = 0.f;
  for (int h = lane; h < kH; h += 32) {
    float xv = xr[h];
    a0 += xv * rw[0 * kH + h];
    a1 += xv * rw[1 * kH + h];
    a2 += xv * rw[2 * kH + h];
    a3 += xv * rw[3 * kH + h];
  }
  for (int off = 16; off; off >>= 1) {
    a0 += __shfl_xor(a0, off, 32);
    a1 += __shfl_xor(a1, off, 32);
    a2 += __shfl_xor(a2, off, 32);
    a3 += __shfl_xor(a3, off, 32);
  }
  if (lane == 0) {
    float lg[kE] = {a0, a1, a2, a3};
    float mx = lg[0];
    for (int e = 1; e < kE; ++e) mx = fmaxf(mx, lg[e]);
    float ex[kE], se = 0.f;
    for (int e = 0; e < kE; ++e) { ex[e] = __expf(lg[e] - mx); se += ex[e]; }
    float p[kE];
    for (int e = 0; e < kE; ++e) p[e] = ex[e] / se;
    int i1 = 0;
    for (int e = 1; e < kE; ++e) if (p[e] > p[i1]) i1 = e;
    int i2 = (i1 == 0) ? 1 : 0;
    for (int e = 0; e < kE; ++e)
      if (e != i1 && e != i2 && p[e] > p[i2]) i2 = e;
    float s = p[i1] + p[i2];
    for (int e = 0; e < kE; ++e)
      wf[(size_t)tok * kE + e] = (e == i1 || e == i2) ? p[e] / s : 0.f;
  }
}

// ---------------------------------------------------------------- fused MoE
// Block: 256 threads = 8 waves, MT=32 tokens. x tile staged once in LDS
// (fused fp32->bf16). Per expert, stream I in 128-wide chunks:
//   phase1: gate/up GEMM over K=H (A from LDS, B streamed from L2-resident
//           bf16 weights), silu(g)*u*w_tok -> bf16 LDS h tile
//   phase2: down GEMM over K=I-chunk into persistent fp32 accumulators.
__global__ __launch_bounds__(256)
void moe_k(const float* __restrict__ x, const u16* __restrict__ gwb,
           const u16* __restrict__ uwb, const u16* __restrict__ dwb,
           const float* __restrict__ wf, float* __restrict__ out) {
  __shared__ u16 x_lds[MT * XSTR];  // 57,856 B padded bf16 x tile
  __shared__ u16 h_lds[MT * HSTR];  // 8,704 B padded bf16 activation tile
  __shared__ float wv[MT];          // per-token routing weight for expert e

  const int tid  = threadIdx.x;
  const int wave = tid >> 5;
  const int lane = tid & 31;
  const int l16  = lane & 15;
  const int hi   = lane >> 4;       // lane half (ISA 16-bit frag layout)
  const size_t tok0 = (size_t)blockIdx.x * MT;

  // ---- stage x tile: fp32 global -> bf16 LDS (RNE), 4 elems per iteration
  for (int idx = tid; idx < MT * (kH / 4); idx += 256) {
    const int row = idx / (kH / 4);
    const int col = (idx % (kH / 4)) * 4;
    float4 f = *reinterpret_cast<const float4*>(x + (tok0 + row) * kH + col);
    u16 o[4] = {f2bf(f.x), f2bf(f.y), f2bf(f.z), f2bf(f.w)};
    *reinterpret_cast<uint2*>(&x_lds[row * XSTR + col]) =
        *reinterpret_cast<const uint2*>(o);
  }

  const v8f vzero = {0.f, 0.f, 0.f, 0.f, 0.f, 0.f, 0.f, 0.f};
  v8f oacc[2][7];
#pragma unroll
  for (int mt = 0; mt < 2; ++mt)
#pragma unroll
    for (int t = 0; t < 7; ++t) oacc[mt][t] = vzero;

  const int mtile = wave >> 2;              // phase1: M-tile of this wave
  const int it0   = (wave & 3) * 2;         // phase1: first of 2 I-tiles

  for (int e = 0; e < kE; ++e) {
    __syncthreads();                        // x staged / prior h_lds reads done
    if (tid < MT) wv[tid] = wf[(tok0 + tid) * kE + e];
    __syncthreads();

    for (int ic = 0; ic < kI / ICH; ++ic) { // 38 chunks
      // ---------------- phase 1: gate & up for two 16x16 (M x I) tiles
      v8f g0 = vzero, g1 = vzero, u0 = vzero, u1 = vzero;
      const u16* xrow = x_lds + ((size_t)mtile * 16 + l16) * XSTR;
      const size_t i0 = (size_t)e * kI + (size_t)ic * ICH + (size_t)it0 * 16;
      const u16* gr0 = gwb + (i0 + l16) * kH;
      const u16* gr1 = gwb + (i0 + 16 + l16) * kH;
      const u16* ur0 = uwb + (i0 + l16) * kH;
      const u16* ur1 = uwb + (i0 + 16 + l16) * kH;

      // speculative prefetch of next chunk's weight rows (global_prefetch_b8)
      if (ic + 1 < kI / ICH) {
        __builtin_prefetch(gr0 + (size_t)ICH * kH, 0, 1);
        __builtin_prefetch(gr1 + (size_t)ICH * kH, 0, 1);
        __builtin_prefetch(ur0 + (size_t)ICH * kH, 0, 1);
        __builtin_prefetch(ur1 + (size_t)ICH * kH, 0, 1);
        __builtin_prefetch(dwb + ((size_t)e * kH + wave * 112 + l16) * kI +
                               (size_t)(ic + 1) * ICH, 0, 1);
      }

      for (int kk = 0; kk < kH; kk += 32) {
        FragU a, bg0, bg1, bu0, bu1;
        // A (16x32 bf16) from LDS: 32B contiguous K per lane half, two ds_load_b128
        a.q[0] = *reinterpret_cast<const uint4*>(xrow + kk + hi * 16);
        a.q[1] = *reinterpret_cast<const uint4*>(xrow + kk + hi * 16 + 8);
        // B (32x16 bf16): lane = column (I row), 32B contiguous K per lane
        bg0.q[0] = *reinterpret_cast<const uint4*>(gr0 + kk + hi * 16);
        bg0.q[1] = *reinterpret_cast<const uint4*>(gr0 + kk + hi * 16 + 8);
        bg1.q[0] = *reinterpret_cast<const uint4*>(gr1 + kk + hi * 16);
        bg1.q[1] = *reinterpret_cast<const uint4*>(gr1 + kk + hi * 16 + 8);
        bu0.q[0] = *reinterpret_cast<const uint4*>(ur0 + kk + hi * 16);
        bu0.q[1] = *reinterpret_cast<const uint4*>(ur0 + kk + hi * 16 + 8);
        bu1.q[0] = *reinterpret_cast<const uint4*>(ur1 + kk + hi * 16);
        bu1.q[1] = *reinterpret_cast<const uint4*>(ur1 + kk + hi * 16 + 8);
        g0 = wmma_bf16(a, bg0, g0);
        g1 = wmma_bf16(a, bg1, g1);
        u0 = wmma_bf16(a, bu0, u0);
        u1 = wmma_bf16(a, bu1, u1);
      }

      __syncthreads();  // previous chunk's phase2 consumed h_lds
      // h = w_tok * silu(gate) * up  -> bf16 LDS tile [MT x ICH]
#pragma unroll
      for (int pair = 0; pair < 2; ++pair) {
        v8f g = pair ? g1 : g0;
        v8f u = pair ? u1 : u0;
        const int itile = it0 + pair;
#pragma unroll
        for (int r = 0; r < 8; ++r) {
          const int tm = mtile * 16 + r + hi * 8;   // C/D layout row
          float gv = g[r];
          float sg = gv / (1.f + __expf(-gv));      // silu
          float hv = wv[tm] * sg * u[r];
          h_lds[tm * HSTR + itile * 16 + l16] = f2bf(hv);
        }
      }
      __syncthreads();

      // ---------------- phase 2: out[32, Hslice] += h @ down_e^T
#pragma unroll
      for (int mt = 0; mt < 2; ++mt) {
        const u16* hrow = h_lds + (mt * 16 + l16) * HSTR;
#pragma unroll
        for (int kk2 = 0; kk2 < ICH; kk2 += 32) {
          FragU a;
          a.q[0] = *reinterpret_cast<const uint4*>(hrow + kk2 + hi * 16);
          a.q[1] = *reinterpret_cast<const uint4*>(hrow + kk2 + hi * 16 + 8);
#pragma unroll
          for (int t = 0; t < 7; ++t) {
            const int h0 = (wave * 7 + t) * 16;
            const u16* db = dwb +
                ((size_t)e * kH + h0 + l16) * kI +
                (size_t)ic * ICH + kk2 + hi * 16;
            FragU b;
            b.q[0] = *reinterpret_cast<const uint4*>(db);
            b.q[1] = *reinterpret_cast<const uint4*>(db + 8);
            oacc[mt][t] = wmma_bf16(a, b, oacc[mt][t]);
          }
        }
      }
    }  // ic
  }    // e

  // single deterministic store of the full [MT x H] tile
#pragma unroll
  for (int mt = 0; mt < 2; ++mt)
#pragma unroll
    for (int t = 0; t < 7; ++t) {
      const size_t col = (size_t)(wave * 7 + t) * 16 + l16;
#pragma unroll
      for (int r = 0; r < 8; ++r) {
        const size_t row = tok0 + mt * 16 + r + hi * 8;
        out[row * kH + col] = oacc[mt][t][r];
      }
    }
}

// ---------------------------------------------------------------- launch
extern "C" void kernel_launch(void* const* d_in, const int* in_sizes, int n_in,
                              void* d_out, int out_size, void* d_ws,
                              size_t ws_size, hipStream_t stream) {
  const float* x  = (const float*)d_in[0];   // [B,S,H]
  const float* rw = (const float*)d_in[1];   // [E,H]
  const float* gw = (const float*)d_in[2];   // [E,I,H]
  const float* uw = (const float*)d_in[3];   // [E,I,H]
  const float* dw = (const float*)d_in[4];   // [E,H,I]
  float* out = (float*)d_out;

  char* ws = (char*)d_ws;
  u16* gwb = (u16*)(ws + OFF_GW);
  u16* uwb = (u16*)(ws + OFF_UW);
  u16* dwb = (u16*)(ws + OFF_DW);
  float* wf = (float*)(ws + OFF_WF);

  const size_t nW = kWElems;                 // per weight tensor
  const int cvtThreads = 256;
  const int cvtBlksW = (int)(nW / (4 * (size_t)cvtThreads));  // exact divide

  cvt_f32_bf16<<<cvtBlksW, cvtThreads, 0, stream>>>(gw, gwb, nW);
  cvt_f32_bf16<<<cvtBlksW, cvtThreads, 0, stream>>>(uw, uwb, nW);
  cvt_f32_bf16<<<cvtBlksW, cvtThreads, 0, stream>>>(dw, dwb, nW);

  router_k<<<kTok / 4, 128, 0, stream>>>(x, rw, wf);

  moe_k<<<kTok / MT, 256, 0, stream>>>(x, gwb, uwb, dwb, wf, out);
}
